// RoIHead_86835648791317
// MI455X (gfx1250) — matrix operand order repaired and checked by reading, not compiled, not run on hardware
//
#include <hip/hip_runtime.h>
#include <hip/hip_bf16.h>

// ---------- problem constants ----------
constexpr int ROIS = 128;          // K rois
constexpr int CIN  = 512;          // channels
constexpr int HH   = 38;
constexpr int WW   = 50;
constexpr int RSZ  = 7;            // pooled size
constexpr int HWIN = 7;
constexpr int WWIN = 9;
constexpr int FEAT = RSZ * RSZ * CIN;   // 25088
constexpr int HID  = 4096;
constexpr int NCLS = 21;
constexpr int NLOC = NCLS * 4;          // 84
constexpr float INV_SCALE = 1.0f / 16.0f;

typedef __bf16 v16bf __attribute__((ext_vector_type(16)));
typedef float  v8f   __attribute__((ext_vector_type(8)));
typedef unsigned u32x4 __attribute__((ext_vector_type(4)));
typedef int      i32x8 __attribute__((ext_vector_type(8)));
typedef int      i32x4 __attribute__((ext_vector_type(4)));

#if defined(__has_builtin)
#if __has_builtin(__builtin_amdgcn_tensor_load_to_lds)
#define HAVE_TDM 1
#endif
#endif

// pack two fp32 -> packed bf16 pair (RNE-ish), low half = first element (even k)
__device__ __forceinline__ unsigned pack_bf16(float lo, float hi) {
    unsigned ulo = __builtin_bit_cast(unsigned, lo);
    unsigned uhi = __builtin_bit_cast(unsigned, hi);
    ulo = ulo + 0x7FFFu + ((ulo >> 16) & 1u);
    uhi = uhi + 0x7FFFu + ((uhi >> 16) & 1u);
    return (ulo >> 16) | (uhi & 0xFFFF0000u);
}

#ifdef HAVE_TDM
// Issue a TDM DMA: copy a 32x32 fp32 tile (row stride N elements) from global
// memory into LDS at byte offset lds_off. D# packing per CDNA5 ISA 8.3/8.4.
__device__ __forceinline__ void tdm_load_tile32x32(const float* gsrc,
                                                   unsigned lds_off,
                                                   int N_, int rows) {
    unsigned long long ga = (unsigned long long)(uintptr_t)gsrc;
    u32x4 g0;
    g0[0] = 1u;                                   // count=1, user mode, no gather
    g0[1] = lds_off;                              // lds_addr [63:32]
    g0[2] = (unsigned)(ga & 0xFFFFFFFFu);         // global_addr [95:64]
    g0[3] = (unsigned)((ga >> 32) & 0x01FFFFFFu)  // global_addr [120:96]
            | (2u << 30);                         // type=2 ("image") [127:126]
    i32x8 g1;
    g1[0] = (int)(2u << 16);                      // data_size=4B; wg_mask=0
    g1[1] = (int)(((unsigned)N_ & 0xFFFFu) << 16);          // tensor_dim0 lo16
    g1[2] = (int)((((unsigned)N_ >> 16) & 0xFFFFu)          // tensor_dim0 hi16
                  | (((unsigned)rows & 0xFFFFu) << 16));    // tensor_dim1 lo16
    g1[3] = (int)((((unsigned)rows >> 16) & 0xFFFFu)        // tensor_dim1 hi16
                  | (32u << 16));                           // tile_dim0 = 32
    g1[4] = 32;                                   // tile_dim1=32, tile_dim2=0
    g1[5] = N_;                                   // tensor_dim0_stride lo32
    g1[6] = 0;                                    // stride hi bits / dim1_stride lo
    g1[7] = 0;                                    // tensor_dim1_stride = 0 (2D)
    i32x4 z4 = {0, 0, 0, 0};
#if __clang_major__ >= 23
    i32x8 z8 = {0, 0, 0, 0, 0, 0, 0, 0};
    __builtin_amdgcn_tensor_load_to_lds(g0, g1, z4, z4, z8, 0);
#else
    __builtin_amdgcn_tensor_load_to_lds(g0, g1, z4, z4, 0);
#endif
}
#endif

// ====================================================================
// Kernel 1: RoI max pool -> x as bf16, flattened index = c*49 + ph*7 + pw
// grid: (128, 49), block: 256
// ====================================================================
__global__ void roi_pool_bf16(const float* __restrict__ feat,
                              const float* __restrict__ rois,
                              __hip_bfloat16* __restrict__ xbf) {
    const int k  = blockIdx.x;
    const int p  = blockIdx.y;
    const int ph = p / RSZ;
    const int pw = p % RSZ;

    const float x1 = rois[k * 5 + 1], y1 = rois[k * 5 + 2];
    const float x2 = rois[k * 5 + 3], y2 = rois[k * 5 + 4];
    const int sx = (int)rintf(x1 * INV_SCALE);
    const int sy = (int)rintf(y1 * INV_SCALE);
    const int ex = (int)rintf(x2 * INV_SCALE);
    const int ey = (int)rintf(y2 * INV_SCALE);
    const float roi_w = fmaxf((float)(ex - sx + 1), 1.0f);
    const float roi_h = fmaxf((float)(ey - sy + 1), 1.0f);

    int hstart = min(max((int)floorf((float)ph * roi_h / (float)RSZ) + sy, 0), HH);
    int hend   = min(max((int)ceilf((float)(ph + 1) * roi_h / (float)RSZ) + sy, 0), HH);
    int wstart = min(max((int)floorf((float)pw * roi_w / (float)RSZ) + sx, 0), WW);
    int wend   = min(max((int)ceilf((float)(pw + 1) * roi_w / (float)RSZ) + sx, 0), WW);

    const int h1e = min(hend, hstart + HWIN);   // reference caps window at HWIN
    const int w1e = min(wend, wstart + WWIN);   // and WWIN
    const bool empty = (hend <= hstart) || (wend <= wstart);

    for (int c = threadIdx.x; c < CIN; c += blockDim.x) {
        float m = -__builtin_inff();
        const float* fc = feat + (size_t)c * HH * WW;
        for (int y = hstart; y < h1e; ++y) {
            const float* fr = fc + y * WW;
            for (int x = wstart; x < w1e; ++x)
                m = fmaxf(m, fr[x]);
        }
        if (empty) m = 0.0f;
        xbf[(size_t)k * FEAT + c * (RSZ * RSZ) + ph * RSZ + pw] = __float2bfloat16(m);
    }
}

// ====================================================================
// Kernel 2: H(128xN) = act( Abf(128xK) @ Wf32(KxN) + bias ), bf16 out
// block = 256 threads = 8 waves; wave w owns M-tile [16w,16w+16)
// blockIdx.x selects a 32-column N slice; loop K in steps of 32.
// Weight tiles streamed via TDM (tensor_load_to_lds) into double-buffered
// fp32 LDS tiles, converted to packed-bf16 WMMA B-fragment order, then
// consumed by v_wmma_f32_16x16x32_bf16.
// ====================================================================
template <int K, int N, bool RELU>
__global__ void __launch_bounds__(256, 2)
gemm_bf16_wmma(const __hip_bfloat16* __restrict__ A,   // 128 x K bf16 row-major
               const float* __restrict__ Wt,           // K x N fp32 row-major
               const float* __restrict__ bias,         // N
               __hip_bfloat16* __restrict__ Out) {     // 128 x N bf16
#ifdef HAVE_TDM
    __shared__ float    ldsW[2][32 * 32];   // double-buffered fp32 weight tiles
#endif
    // ldsB[c][rp] : packed bf16 pair of rows (2rp, 2rp+1) of tile column c
    __shared__ unsigned ldsB[32][17];       // pitch 17 -> conflict-free reads

    const int tid  = threadIdx.x;
    const int wave = tid >> 5;          // 0..7  == M-tile
    const int lane = tid & 31;
    const int g    = lane >> 4;         // lane half
    const int l15  = lane & 15;
    const int n0   = blockIdx.x * 32;

    // staging coordinates: one column, two row-pairs per thread
    const int cst = tid & 31;
    const int rp  = tid >> 5;           // 0..7 ; also handles rp+8

    v8f acc0 = {};
    v8f acc1 = {};

#ifdef HAVE_TDM
    // prologue: DMA first weight tile into buffer 0
    if (wave == 0) {
        tdm_load_tile32x32(Wt + (size_t)0 * N + n0,
                           (unsigned)(uintptr_t)&ldsW[0][0], N, K);
    }
#endif

    for (int kk = 0; kk < K; kk += 32) {
        const int buf = (kk >> 5) & 1;

#ifdef HAVE_TDM
        // current tile's DMA complete (wave 0 tracks TENSORcnt), publish via barrier
        if (wave == 0) __builtin_amdgcn_s_wait_tensorcnt(0);
        __syncthreads();
        // kick off next tile's DMA into the other buffer
        if (wave == 0 && (kk + 32) < K) {
            tdm_load_tile32x32(Wt + (size_t)(kk + 32) * N + n0,
                               (unsigned)(uintptr_t)&ldsW[buf ^ 1][0], N, K - kk - 32);
        }
        // ---- convert fp32 tile -> packed bf16 pairs (WMMA B layout) ----
        {
            float a0 = ldsW[buf][(2 * rp) * 32 + cst];
            float a1 = ldsW[buf][(2 * rp + 1) * 32 + cst];
            float a2 = ldsW[buf][(2 * rp + 16) * 32 + cst];
            float a3 = ldsW[buf][(2 * rp + 17) * 32 + cst];
            ldsB[cst][rp]     = pack_bf16(a0, a1);
            ldsB[cst][rp + 8] = pack_bf16(a2, a3);
        }
#else
        (void)buf;
        // fallback: stage 32x32 fp32 weight tile straight from global
        {
            const float* wp = Wt + (size_t)kk * N + n0 + cst;
            float a0 = wp[(size_t)(2 * rp) * N];
            float a1 = wp[(size_t)(2 * rp + 1) * N];
            float a2 = wp[(size_t)(2 * rp + 16) * N];
            float a3 = wp[(size_t)(2 * rp + 17) * N];
            __builtin_prefetch(wp + (size_t)32 * N, 0, 1);
            ldsB[cst][rp]     = pack_bf16(a0, a1);
            ldsB[cst][rp + 8] = pack_bf16(a2, a3);
        }
#endif
        __syncthreads();

        // ---- A fragment: 16x32 bf16, contiguous layout (lanes 0-15: k 0-15,
        //      lanes 16-31: k 16-31), 2-per-VGPR consecutive k ----
        union { uint4 q[2]; v16bf v; } afr;
        {
            const uint4* ap = (const uint4*)(A + (size_t)(16 * wave + l15) * K + kk + 16 * g);
            afr.q[0] = ap[0];   // k = 16g .. 16g+7
            afr.q[1] = ap[1];   // k = 16g+8 .. 16g+15
        }

        // ---- B fragments from LDS (column n = l15 (+16), rows 16g..16g+15) ----
        union { unsigned u[8]; v16bf v; } bf0, bf1;
#pragma unroll
        for (int t = 0; t < 8; ++t) {
            bf0.u[t] = ldsB[l15][g * 8 + t];
            bf1.u[t] = ldsB[16 + l15][g * 8 + t];
        }

        acc0 = __builtin_amdgcn_wmma_f32_16x16x32_bf16(false, afr.v, false, bf0.v,
                                                       (short)0, acc0, false, false);
        acc1 = __builtin_amdgcn_wmma_f32_16x16x32_bf16(false, afr.v, false, bf1.v,
                                                       (short)0, acc1, false, false);
        __syncthreads();
    }

    // ---- epilogue: bias + (ReLU) + bf16 store ----
    const float b0 = bias[n0 + l15];
    const float b1 = bias[n0 + 16 + l15];
#pragma unroll
    for (int r = 0; r < 8; ++r) {
        const int m = 16 * wave + r + 8 * g;     // C/D layout: lanes16-31 -> M+8
        float o0 = acc0[r] + b0;
        float o1 = acc1[r] + b1;
        if (RELU) { o0 = fmaxf(o0, 0.0f); o1 = fmaxf(o1, 0.0f); }
        Out[(size_t)m * N + n0 + l15]      = __float2bfloat16(o0);
        Out[(size_t)m * N + n0 + 16 + l15] = __float2bfloat16(o1);
    }
}

// ====================================================================
// Kernel 3: tiny heads. locs = h@Wr+br (128x84), scores = h@Wc+bc (128x21)
// one block per roi; h row cached in LDS.
// ====================================================================
__global__ void __launch_bounds__(128)
heads_kernel(const __hip_bfloat16* __restrict__ h,    // 128 x 4096 bf16
             const float* __restrict__ Wr, const float* __restrict__ br,
             const float* __restrict__ Wc, const float* __restrict__ bc,
             float* __restrict__ out) {               // [128*84 | 128*21]
    __shared__ float hrow[HID];
    const int k = blockIdx.x;
    for (int i = threadIdx.x; i < HID; i += blockDim.x)
        hrow[i] = __bfloat162float(h[(size_t)k * HID + i]);
    __syncthreads();

    const int j = threadIdx.x;
    if (j < NLOC) {
        float acc = br[j];
        for (int i = 0; i < HID; ++i)
            acc = fmaf(hrow[i], Wr[(size_t)i * NLOC + j], acc);
        out[(size_t)k * NLOC + j] = acc;
    } else if (j < NLOC + NCLS) {
        const int c = j - NLOC;
        float acc = bc[c];
        for (int i = 0; i < HID; ++i)
            acc = fmaf(hrow[i], Wc[(size_t)i * NCLS + c], acc);
        out[(size_t)ROIS * NLOC + (size_t)k * NCLS + c] = acc;
    }
}

// ====================================================================
// launcher
// ====================================================================
extern "C" void kernel_launch(void* const* d_in, const int* in_sizes, int n_in,
                              void* d_out, int out_size, void* d_ws, size_t ws_size,
                              hipStream_t stream) {
    const float* features = (const float*)d_in[0];
    const float* rois     = (const float*)d_in[1];
    const float* W1       = (const float*)d_in[2];
    const float* b1       = (const float*)d_in[3];
    const float* W2       = (const float*)d_in[4];
    const float* b2       = (const float*)d_in[5];
    const float* Wr       = (const float*)d_in[6];
    const float* br       = (const float*)d_in[7];
    const float* Wc       = (const float*)d_in[8];
    const float* bc       = (const float*)d_in[9];
    float* out            = (float*)d_out;

    char* ws = (char*)d_ws;
    const size_t XBF_BYTES = (size_t)ROIS * FEAT * 2;      // 6,422,528 (256B aligned)
    const size_t H_BYTES   = (size_t)ROIS * HID  * 2;      // 1,048,576
    __hip_bfloat16* xbf = (__hip_bfloat16*)(ws);
    __hip_bfloat16* h1  = (__hip_bfloat16*)(ws + XBF_BYTES);
    __hip_bfloat16* h2  = (__hip_bfloat16*)(ws + XBF_BYTES + H_BYTES);

    // 1) RoI max pool -> bf16 activations
    roi_pool_bf16<<<dim3(ROIS, RSZ * RSZ), 256, 0, stream>>>(features, rois, xbf);

    // 2) FC1: (128 x 25088) @ (25088 x 4096), ReLU   -- streams 411MB of W1
    gemm_bf16_wmma<FEAT, HID, true><<<HID / 32, 256, 0, stream>>>(xbf, W1, b1, h1);

    // 3) FC2: (128 x 4096) @ (4096 x 4096), ReLU
    gemm_bf16_wmma<HID, HID, true><<<HID / 32, 256, 0, stream>>>(h1, W2, b2, h2);

    // 4) heads -> fp32 output (locs then scores)
    heads_kernel<<<ROIS, 128, 0, stream>>>(h2, Wr, br, Wc, bc, out);
}
